// CrossAttenGenerator_21294447854062
// MI455X (gfx1250) — compile-verified, matrix operand order checked by reading, not run on hardware
//
#include <hip/hip_runtime.h>
#include <hip/hip_bf16.h>
#include <cstdint>
#include <cstddef>

// ---------------------------------------------------------------------------
// CDNA5 (gfx1250) implementation of the CrossAttenGenerator forward pass.
// Strategy: every conv / linear / attention matmul is a bf16 WMMA GEMM
// (v_wmma_f32_16x16x32_bf16), convs via implicit im2col.  GEMM tiles are
// staged into LDS by the Tensor Data Mover (tensor_load_to_lds, TENSORcnt)
// with double-buffering: the DMA for tile k+1 overlaps the WMMA math on
// tile k (wait tensorcnt<=2 leaves the just-issued pair in flight).  TDM
// LDS-padding (1 dword per 16) de-conflicts LDS banks on fragment reads.
// ---------------------------------------------------------------------------

typedef __attribute__((ext_vector_type(16))) __bf16    v16bf;
typedef __attribute__((ext_vector_type(8)))  float     v8f;
typedef __attribute__((ext_vector_type(4)))  unsigned int u32x4;
typedef __attribute__((ext_vector_type(8)))  int       i32x8;
typedef __attribute__((ext_vector_type(4)))  int       i32x4;

__device__ __forceinline__ uint16_t f32_to_bf16(float f) {
  union { float f; uint32_t u; } v; v.f = f;
  uint32_t r = v.u + 0x7FFFu + ((v.u >> 16) & 1u);   // round-to-nearest-even
  return (uint16_t)(r >> 16);
}
__device__ __forceinline__ float bf16_to_f32(uint16_t h) {
  union { uint32_t u; float f; } v; v.u = ((uint32_t)h) << 16; return v.f;
}

// ---------------------------------------------------------------------------
// GEMM:  C[M x N] = act( alpha * (A[M x K] * Bt[N x K]^T) * scale + bias + add )
// A, Bt bf16 row-major (K contiguous).  flags: 1=relu 2=scale 4=bias 8=add.
// ---------------------------------------------------------------------------
#define USE_TDM 1
#define GEMM_LDSW 34   // 32 bf16 + 2 pad = 17 dwords per LDS row

union BF16Frag { v16bf v; uint32_t u[8]; };

// TDM 2D tile load: 128 rows x 32 bf16, dest LDS row stride 17 dwords via
// pad_enable (pad_interval=3 -> every 16 dwords, pad_amount=0 -> 1 dword).
// tensor_dim0/1 are "remaining" extents so OOB rows/cols zero-fill.
__device__ __forceinline__ void tdm_load_tile_bf16(uint32_t lds_off,
                                                   const uint16_t* gptr,
                                                   uint32_t rem_k, uint32_t rem_m,
                                                   uint64_t row_stride_elems) {
#if USE_TDM
  uint64_t ga = (uint64_t)(uintptr_t)gptr;
  u32x4 g0;
  g0[0] = 1u;                                            // count=1, user descriptor
  g0[1] = lds_off;                                       // lds_addr (bytes)
  g0[2] = (uint32_t)ga;                                  // global_addr[31:0]
  g0[3] = (uint32_t)((ga >> 32) & 0x1FFFFFFu) | (2u << 30); // addr[56:32] | type=2
  i32x8 g1;
  g1[0] = (int)((1u << 16) | (1u << 20) | (3u << 22));   // data_size=2B, pad_en, ivl=16dw, amt=1dw
  g1[1] = (int)((rem_k & 0xFFFFu) << 16);                // [47:32]=0 barrier, [63:48]=dim0.lo
  g1[2] = (int)((rem_k >> 16) | ((rem_m & 0xFFFFu) << 16));
  g1[3] = (int)((rem_m >> 16) | (32u << 16));            // tile_dim0 = 32
  g1[4] = (int)(128u);                                   // tile_dim1 = 128, tile_dim2 = 0
  g1[5] = (int)(uint32_t)(row_stride_elems & 0xFFFFFFFFu);
  g1[6] = (int)(uint32_t)((row_stride_elems >> 32) & 0xFFFFu);
  g1[7] = 0;
  i32x4 z4 = {0, 0, 0, 0};
  i32x8 z8 = {0, 0, 0, 0, 0, 0, 0, 0};
  __builtin_amdgcn_tensor_load_to_lds(g0, g1, z4, z4, z8, 0);
#endif
}

__global__ __launch_bounds__(256)
void k_gemm_bf16(const uint16_t* __restrict__ A, const uint16_t* __restrict__ Bt,
                 float* __restrict__ C, uint16_t* __restrict__ Cbf,
                 const float* __restrict__ scale, const float* __restrict__ bias,
                 const float* __restrict__ addsrc,
                 int M, int N, int K, int lda, int ldb, int ldc,
                 float alpha, int flags) {
  __shared__ uint16_t As[2][128 * GEMM_LDSW];
  __shared__ uint16_t Bs[2][128 * GEMM_LDSW];
  const int tid  = threadIdx.x;
  const int wave = tid >> 5, lane = tid & 31;
  const int half = lane >> 4, lrow = lane & 15;
  const int wm = wave >> 2, wn = wave & 3;     // 2 x 4 wave grid, 64x32 per wave
  const int row0 = blockIdx.y * 128, col0 = blockIdx.x * 128;

  const v8f vzero = {0.f, 0.f, 0.f, 0.f, 0.f, 0.f, 0.f, 0.f};
  v8f acc[4][2];
#pragma unroll
  for (int i = 0; i < 4; i++)
#pragma unroll
    for (int j = 0; j < 2; j++) acc[i][j] = vzero;

  const uint32_t remM = (uint32_t)(M > row0 ? M - row0 : 0);
  const uint32_t remN = (uint32_t)(N > col0 ? N - col0 : 0);

#if USE_TDM
  if (wave == 0) {   // uniform per-wave branch: one wave drives the DMA engine
    tdm_load_tile_bf16((uint32_t)(uintptr_t)&As[0][0],
                       A + (size_t)row0 * lda, (uint32_t)K, remM, (uint64_t)lda);
    tdm_load_tile_bf16((uint32_t)(uintptr_t)&Bs[0][0],
                       Bt + (size_t)col0 * ldb, (uint32_t)K, remN, (uint64_t)ldb);
  }
#endif

  int cur = 0;
  for (int k0 = 0; k0 < K; k0 += 32) {
    const bool hasNext = (k0 + 32) < K;
#if USE_TDM
    if (wave == 0) {
      if (hasNext) {
        // Prefetch next K-tile into the other buffer, then wait only for the
        // *previous* pair (<=2 outstanding): DMA overlaps this tile's math.
        const int kn = k0 + 32;
        tdm_load_tile_bf16((uint32_t)(uintptr_t)&As[cur ^ 1][0],
                           A + (size_t)row0 * lda + kn, (uint32_t)(K - kn), remM,
                           (uint64_t)lda);
        tdm_load_tile_bf16((uint32_t)(uintptr_t)&Bs[cur ^ 1][0],
                           Bt + (size_t)col0 * ldb + kn, (uint32_t)(K - kn), remN,
                           (uint64_t)ldb);
        __builtin_amdgcn_s_wait_tensorcnt(2);
      } else {
        __builtin_amdgcn_s_wait_tensorcnt(0);
      }
    }
#else
    for (int i = tid; i < 128 * 32; i += 256) {
      int r = i >> 5, c = i & 31;
      int gr = row0 + r, gc = k0 + c;
      As[cur][r * GEMM_LDSW + c] = (gr < M && gc < K) ? A[(size_t)gr * lda + gc] : (uint16_t)0;
      gr = col0 + r;
      Bs[cur][r * GEMM_LDSW + c] = (gr < N && gc < K) ? Bt[(size_t)gr * ldb + gc] : (uint16_t)0;
    }
#endif
    __syncthreads();

    // Fragment gather per documented 16-bit A layout: lane<16 -> K 0-7,16-23;
    // lane>=16 -> +8.  Pair i maps to k = half*8 + (i<4 ? 2i : 16+2(i-4)).
    BF16Frag af[4], bf[2];
#pragma unroll
    for (int i = 0; i < 8; i++) {
      const int kk = half * 8 + ((i & 4) ? (16 + ((i & 3) << 1)) : (i << 1));
#pragma unroll
      for (int mt = 0; mt < 4; mt++)
        af[mt].u[i] = *(const uint32_t*)&As[cur][(wm * 64 + mt * 16 + lrow) * GEMM_LDSW + kk];
#pragma unroll
      for (int nt = 0; nt < 2; nt++)
        bf[nt].u[i] = *(const uint32_t*)&Bs[cur][(wn * 32 + nt * 16 + lrow) * GEMM_LDSW + kk];
    }
#pragma unroll
    for (int mt = 0; mt < 4; mt++)
#pragma unroll
      for (int nt = 0; nt < 2; nt++)
        acc[mt][nt] = __builtin_amdgcn_wmma_f32_16x16x32_bf16(
            false, af[mt].v, false, bf[nt].v, (short)0, acc[mt][nt], false, false);
    __syncthreads();
    cur ^= 1;
  }

  // Epilogue: C/D layout -> row = base + half*8 + r, col = base + (lane&15).
#pragma unroll
  for (int mt = 0; mt < 4; mt++) {
#pragma unroll
    for (int nt = 0; nt < 2; nt++) {
      const int col = col0 + wn * 32 + nt * 16 + lrow;
      if (col >= N) continue;
      const float as = alpha * ((flags & 2) ? scale[col] : 1.f);
      const float bi = (flags & 4) ? bias[col] : 0.f;
#pragma unroll
      for (int r = 0; r < 8; r++) {
        const int row = row0 + wm * 64 + mt * 16 + half * 8 + r;
        if (row >= M) continue;
        float y = acc[mt][nt][r] * as + bi;
        if (flags & 8) y += addsrc[(size_t)row * ldc + col];
        if (flags & 1) y = fmaxf(y, 0.f);
        if (C)   C[(size_t)row * ldc + col] = y;
        if (Cbf) Cbf[(size_t)row * ldc + col] = f32_to_bf16(y);
      }
    }
  }
}

// ---------------------------------------------------------------------------
// Support kernels (VALU / vmem; all cheap relative to the GEMMs)
// ---------------------------------------------------------------------------
__global__ void k_spectral_z(const float* __restrict__ cond, const float* __restrict__ snw,
                             const float* __restrict__ snu, float* __restrict__ z) {
  __shared__ float sh[512];
  __shared__ float svv[512];
  __shared__ float wv[16];
  __shared__ float s_sig;
  int t = threadIdx.x;
  float vj = 0.f;
  for (int i = 0; i < 16; i++) vj += snw[i * 512 + t] * snu[i];
  sh[t] = vj * vj; __syncthreads();
  for (int s = 256; s > 0; s >>= 1) { if (t < s) sh[t] += sh[t + s]; __syncthreads(); }
  float v = vj / (sqrtf(sh[0]) + 1e-12f);
  svv[t] = v; __syncthreads();
  if (t < 16) { float a = 0.f; for (int j = 0; j < 512; j++) a += snw[t * 512 + j] * svv[j]; wv[t] = a; }
  __syncthreads();
  if (t == 0) { float s = 0.f; for (int i = 0; i < 16; i++) s += wv[i] * wv[i]; s_sig = sqrtf(s); }
  __syncthreads();
  float nrm = s_sig + 1e-12f;
  float sigma = s_sig * s_sig / nrm;            // u_hat . (W v) with u_hat = Wv/||Wv||
  if (t < 64) {
    int b = t >> 4, i = t & 15;
    float a = 0.f;
    for (int j = 0; j < 512; j++) a += cond[b * 512 + j] * snw[i * 512 + j];
    z[b * 16 + i] = a / sigma;
  }
}

__global__ void k_nchw_to_nhwc_bf16(const float* __restrict__ x, uint16_t* __restrict__ out,
                                    int Bn, int C, int H, int W) {
  size_t n = (size_t)Bn * C * H * W;
  size_t idx = (size_t)blockIdx.x * 256 + threadIdx.x;
  if (idx >= n) return;
  int w = (int)(idx % W); size_t r = idx / W;
  int h = (int)(r % H); r /= H;
  int c = (int)(r % C); int b = (int)(r / C);
  out[(((size_t)b * H + h) * W + w) * C + c] = f32_to_bf16(x[idx]);
}

// Weight prep: dst row o, col = i*KH*KW + kh*KW + kw.  flipswap=1 handles the
// convT weights (source layout (I,O,KH,KW), spatially flipped).
__global__ void k_prep_w(const float* __restrict__ src, uint16_t* __restrict__ dst,
                         int O, int I, int KH, int KW, int flipswap) {
  size_t n = (size_t)O * I * KH * KW;
  size_t idx = (size_t)blockIdx.x * 256 + threadIdx.x;
  if (idx >= n) return;
  int kw = (int)(idx % KW); size_t r = idx / KW;
  int kh = (int)(r % KH); r /= KH;
  int i = (int)(r % I); int o = (int)(r / I);
  float v;
  if (!flipswap) v = src[(((size_t)o * I + i) * KH + kh) * KW + kw];
  else           v = src[(((size_t)i * O + o) * KH + (KH - 1 - kh)) * KW + (KW - 1 - kw)];
  dst[idx] = f32_to_bf16(v);
}

__global__ void k_pack_bf16(const float* __restrict__ x, uint16_t* __restrict__ out, size_t n) {
  size_t idx = (size_t)blockIdx.x * 256 + threadIdx.x;
  if (idx < n) out[idx] = f32_to_bf16(x[idx]);
}

// Implicit im2col (one batch image, one row-chunk).  Column order (c,kh,kw)
// matches the flattened OIHW weights.  Handles zero/reflect pad, input
// dilation (for the stride-2 transposed convs) and the appended z-channels.
__global__ void k_im2col(const uint16_t* __restrict__ x, const float* __restrict__ z,
                         uint16_t* __restrict__ out,
                         int b, int H, int W, int C, int NZk, int KH, int KW,
                         int stride, int pad, int dil, int reflect,
                         int OW, int oh0, int Mrows, int Kc) {
  size_t idx = (size_t)blockIdx.x * 256 + threadIdx.x;
  if (idx >= (size_t)Mrows * Kc) return;
  int kcol = (int)(idx % Kc);
  int rr = (int)(idx / Kc);
  int oh = oh0 + rr / OW, ow = rr % OW;
  int khw = KH * KW;
  int c = kcol / khw, rem = kcol % khw;
  int kh = rem / KW, kw = rem % KW;
  int ihl = oh * stride - pad + kh;
  int iwl = ow * stride - pad + kw;
  bool ok = true; int ih = ihl, iw = iwl;
  if (dil > 1) {
    if (ihl < 0 || iwl < 0 || (ihl % dil) != 0 || (iwl % dil) != 0) ok = false;
    ih = ihl / dil; iw = iwl / dil;
  }
  if (ok) {
    if (reflect) {
      if (ih < 0) ih = -ih;
      if (ih >= H) ih = 2 * H - 2 - ih;
      if (iw < 0) iw = -iw;
      if (iw >= W) iw = 2 * W - 2 - iw;
    } else if (ih < 0 || ih >= H || iw < 0 || iw >= W) ok = false;
  }
  uint16_t val = 0;
  if (ok) {
    if (c < C) val = x[(((size_t)b * H + ih) * W + iw) * C + c];
    else       val = f32_to_bf16(z[b * NZk + (c - C)]);
  }
  out[idx] = val;
}

__global__ void k_layernorm_bf16(const float* __restrict__ x, const float* __restrict__ g,
                                 const float* __restrict__ b, uint16_t* __restrict__ out, int C) {
  int row = blockIdx.x, t = threadIdx.x;           // blockDim == C == 256
  __shared__ float sm[256];
  float v = x[(size_t)row * C + t];
  sm[t] = v; __syncthreads();
  for (int s = 128; s > 0; s >>= 1) { if (t < s) sm[t] += sm[t + s]; __syncthreads(); }
  float mean = sm[0] / (float)C; __syncthreads();
  float d = v - mean;
  sm[t] = d * d; __syncthreads();
  for (int s = 128; s > 0; s >>= 1) { if (t < s) sm[t] += sm[t + s]; __syncthreads(); }
  float inv = rsqrtf(sm[0] / (float)C + 1e-5f);
  out[(size_t)row * C + t] = f32_to_bf16(d * inv * g[t] + b[t]);
}

__global__ void k_groupnorm_bf16(const float* __restrict__ x, const float* __restrict__ g,
                                 const float* __restrict__ b, uint16_t* __restrict__ out,
                                 int HW, int C, int cpg) {
  int groups = C / cpg;
  int batch = blockIdx.x / groups, gr = blockIdx.x % groups;
  const float* xp = x + (size_t)batch * HW * C;
  int t = threadIdx.x;
  __shared__ float ss[256], ss2[256];
  float s = 0.f, s2 = 0.f;
  int n = HW * cpg;
  for (int idx = t; idx < n; idx += 256) {
    int tk = idx / cpg, j = idx % cpg;
    float v = xp[(size_t)tk * C + gr * cpg + j];
    s += v; s2 += v * v;
  }
  ss[t] = s; ss2[t] = s2; __syncthreads();
  for (int st = 128; st > 0; st >>= 1) {
    if (t < st) { ss[t] += ss[t + st]; ss2[t] += ss2[t + st]; }
    __syncthreads();
  }
  float mean = ss[0] / (float)n;
  float inv = rsqrtf(ss2[0] / (float)n - mean * mean + 1e-6f);
  for (int idx = t; idx < n; idx += 256) {
    int tk = idx / cpg, j = idx % cpg; int c = gr * cpg + j;
    float v = (xp[(size_t)tk * C + c] - mean) * inv * g[c] + b[c];
    out[(size_t)batch * HW * C + (size_t)tk * C + c] = f32_to_bf16(v);
  }
}

__global__ void k_softmax_bf16(const float* __restrict__ x, uint16_t* __restrict__ out, int ncol) {
  size_t row = blockIdx.x; int t = threadIdx.x;
  const float* xr = x + row * (size_t)ncol;
  __shared__ float sh[256];
  float m = -3.0e38f;
  for (int i = t; i < ncol; i += 256) m = fmaxf(m, xr[i]);
  sh[t] = m; __syncthreads();
  for (int s = 128; s > 0; s >>= 1) { if (t < s) sh[t] = fmaxf(sh[t], sh[t + s]); __syncthreads(); }
  m = sh[0]; __syncthreads();
  float sum = 0.f;
  for (int i = t; i < ncol; i += 256) sum += __expf(xr[i] - m);
  sh[t] = sum; __syncthreads();
  for (int s = 128; s > 0; s >>= 1) { if (t < s) sh[t] += sh[t + s]; __syncthreads(); }
  float inv = 1.f / sh[0];
  for (int i = t; i < ncol; i += 256)
    out[row * (size_t)ncol + i] = f32_to_bf16(__expf(xr[i] - m) * inv);
}

__global__ void k_transpose_bf16(const float* __restrict__ in, uint16_t* __restrict__ out,
                                 int M, int N) {
  size_t idx = (size_t)blockIdx.x * 256 + threadIdx.x;
  if (idx >= (size_t)M * N) return;
  int n = (int)(idx % N); int m = (int)(idx / N);
  out[(size_t)n * M + m] = f32_to_bf16(in[idx]);
}

__global__ void k_geglu_bf16(const float* __restrict__ p, uint16_t* __restrict__ out,
                             size_t M, int half) {
  size_t idx = (size_t)blockIdx.x * 256 + threadIdx.x;
  if (idx >= M * (size_t)half) return;
  size_t m = idx / half; int j = (int)(idx % half);
  float a = p[m * (size_t)(2 * half) + j];
  float g = p[m * (size_t)(2 * half) + half + j];
  float gl = 0.5f * g * (1.f + erff(g * 0.70710678118654752f));
  out[idx] = f32_to_bf16(a * gl);
}

__global__ void k_tanh_blur(const float* __restrict__ x /*NHWC, 3ch*/, float* __restrict__ out,
                            int Bn, int H, int W, const int* __restrict__ epsp) {
  size_t n = (size_t)Bn * 3 * H * W;
  size_t idx = (size_t)blockIdx.x * 256 + threadIdx.x;
  if (idx >= n) return;
  int w = (int)(idx % W); size_t r = idx / W;
  int h = (int)(r % H); r /= H;
  int c = (int)(r % 3); int b = (int)(r / 3);
  const float e1 = 0.60653065971f, e2 = 0.36787944117f;
  const float wnorm = 1.f + 4.f * e1 + 4.f * e2;
  float acc = 0.f;
  for (int dy = -1; dy <= 1; dy++)
    for (int dx = -1; dx <= 1; dx++) {
      int ih = h + dy, iw = w + dx;
      if (ih < 0 || ih >= H || iw < 0 || iw >= W) continue;
      float wt = (dy == 0 && dx == 0) ? 1.f : ((dy == 0 || dx == 0) ? e1 : e2);
      acc += wt * tanhf(x[(((size_t)b * H + ih) * W + iw) * 3 + c]);
    }
  out[idx] = acc / wnorm * (float)(*epsp);
}

// ---------------------------------------------------------------------------
// Host orchestration
// ---------------------------------------------------------------------------
static inline int hmin(int a, int b) { return a < b ? a : b; }

extern "C" void kernel_launch(void* const* d_in, const int* in_sizes, int n_in,
                              void* d_out, int out_size, void* d_ws, size_t ws_size,
                              hipStream_t stream) {
  (void)in_sizes; (void)n_in; (void)out_size; (void)ws_size;
  const float* in_input = (const float*)d_in[0];
  const float* in_cond  = (const float*)d_in[1];
  const float* sn_w = (const float*)d_in[2];
  const float* sn_u = (const float*)d_in[3];
  const float* c1_w = (const float*)d_in[4];
  const float* bn1_g = (const float*)d_in[5];  const float* bn1_b = (const float*)d_in[6];
  const float* c2_w = (const float*)d_in[7];
  const float* bn2_g = (const float*)d_in[8];  const float* bn2_b = (const float*)d_in[9];
  const float* c3_w = (const float*)d_in[10];
  const float* bn3_g = (const float*)d_in[11]; const float* bn3_b = (const float*)d_in[12];
  const float* res_w = (const float*)d_in[13];
  const float* res_g = (const float*)d_in[14]; const float* res_b = (const float*)d_in[15];
  const float* st_gn_g = (const float*)d_in[16]; const float* st_gn_b = (const float*)d_in[17];
  const float* st_pin_w = (const float*)d_in[18]; const float* st_pin_b = (const float*)d_in[19];
  const float* st_q1 = (const float*)d_in[20]; const float* st_k1 = (const float*)d_in[21];
  const float* st_v1 = (const float*)d_in[22]; const float* st_o1w = (const float*)d_in[23];
  const float* st_o1b = (const float*)d_in[24];
  const float* st_q2 = (const float*)d_in[25]; const float* st_k2 = (const float*)d_in[26];
  const float* st_v2 = (const float*)d_in[27]; const float* st_o2w = (const float*)d_in[28];
  const float* st_o2b = (const float*)d_in[29];
  const float* st_ffp_w = (const float*)d_in[30]; const float* st_ffp_b = (const float*)d_in[31];
  const float* st_ffo_w = (const float*)d_in[32]; const float* st_ffo_b = (const float*)d_in[33];
  const float* st_ln1g = (const float*)d_in[34]; const float* st_ln1b = (const float*)d_in[35];
  const float* st_ln2g = (const float*)d_in[36]; const float* st_ln2b = (const float*)d_in[37];
  const float* st_ln3g = (const float*)d_in[38]; const float* st_ln3b = (const float*)d_in[39];
  const float* st_pout_w = (const float*)d_in[40]; const float* st_pout_b = (const float*)d_in[41];
  const float* up1_w = (const float*)d_in[42]; const float* up1_g = (const float*)d_in[43];
  const float* up1_b = (const float*)d_in[44];
  const float* up2_w = (const float*)d_in[45]; const float* up2_g = (const float*)d_in[46];
  const float* up2_b = (const float*)d_in[47];
  const float* cf_w = (const float*)d_in[48]; const float* cf_b = (const float*)d_in[49];
  const int*   epsp = (const int*)d_in[50];

  const int Bn = 4, H0 = 224, W0 = 224, C4 = 256;
  const int HW = 56 * 56;
  const size_t Mtok = (size_t)Bn * HW;            // 12544
  const float bnInv = 0.99999500003749972f;       // 1/sqrt(1+1e-5)

  // ---- bump allocator over d_ws ----
  char* wsBase = (char*)d_ws; size_t wsOff = 0;
  auto alloc = [&](size_t bytes) -> void* {
    void* p = wsBase + wsOff;
    wsOff = (wsOff + bytes + 255) & ~(size_t)255;
    return p;
  };

  float* zvec = (float*)alloc(64 * 4);
  uint16_t* x_in_bf = (uint16_t*)alloc((size_t)Bn * H0 * W0 * 3 * 2);
  uint16_t* w_c1 = (uint16_t*)alloc((size_t)64 * 19 * 49 * 2);
  uint16_t* w_c2 = (uint16_t*)alloc((size_t)128 * 80 * 9 * 2);
  uint16_t* w_c3 = (uint16_t*)alloc((size_t)256 * 144 * 9 * 2);
  uint16_t* w_res[12];
  for (int r = 0; r < 12; r++) w_res[r] = (uint16_t*)alloc((size_t)256 * 2304 * 2);
  uint16_t *w_pin[2], *w_pout[2], *w_q1[2], *w_k1[2], *w_v1[2], *w_o1[2];
  uint16_t *w_q2[2], *w_k2[2], *w_v2[2], *w_o2[2], *w_ffp[2], *w_ffo[2];
  for (int i = 0; i < 2; i++) {
    w_pin[i]  = (uint16_t*)alloc((size_t)256 * 256 * 2);
    w_pout[i] = (uint16_t*)alloc((size_t)256 * 256 * 2);
    w_q1[i]   = (uint16_t*)alloc((size_t)256 * 256 * 2);
    w_k1[i]   = (uint16_t*)alloc((size_t)256 * 256 * 2);
    w_v1[i]   = (uint16_t*)alloc((size_t)256 * 256 * 2);
    w_o1[i]   = (uint16_t*)alloc((size_t)256 * 256 * 2);
    w_q2[i]   = (uint16_t*)alloc((size_t)256 * 256 * 2);
    w_k2[i]   = (uint16_t*)alloc((size_t)256 * 512 * 2);
    w_v2[i]   = (uint16_t*)alloc((size_t)256 * 512 * 2);
    w_o2[i]   = (uint16_t*)alloc((size_t)256 * 256 * 2);
    w_ffp[i]  = (uint16_t*)alloc((size_t)2048 * 256 * 2);
    w_ffo[i]  = (uint16_t*)alloc((size_t)256 * 1024 * 2);
  }
  uint16_t* w_up1 = (uint16_t*)alloc((size_t)128 * 2304 * 2);
  uint16_t* w_up2 = (uint16_t*)alloc((size_t)64 * 1152 * 2);
  uint16_t* w_cf  = (uint16_t*)alloc((size_t)3 * 3136 * 2);

  const size_t actMax = (size_t)Bn * H0 * W0 * 64;          // largest NHWC tensor
  float*    xfA = (float*)alloc(actMax * 4);
  float*    xfB = (float*)alloc(actMax * 4);
  uint16_t* xbA = (uint16_t*)alloc(actMax * 2);
  uint16_t* xbB = (uint16_t*)alloc(actMax * 2);

  uint16_t* h_bf = (uint16_t*)alloc(Mtok * C4 * 2);
  uint16_t* q_bf = (uint16_t*)alloc(Mtok * C4 * 2);
  uint16_t* k_bf = (uint16_t*)alloc(Mtok * C4 * 2);
  float*    v_f  = (float*)alloc(Mtok * C4 * 4);
  uint16_t* vt_bf = (uint16_t*)alloc((size_t)HW * C4 * 2);
  uint16_t* o_bf = (uint16_t*)alloc(Mtok * C4 * 2);
  float*    scoresF = (float*)alloc((size_t)HW * HW * 4);
  uint16_t* probsB  = (uint16_t*)alloc((size_t)HW * HW * 2);
  float*    pF   = (float*)alloc(Mtok * 2048 * 4);
  uint16_t* g_bf = (uint16_t*)alloc(Mtok * 1024 * 2);
  uint16_t* t_bf = (uint16_t*)alloc(Mtok * C4 * 2);
  float*    tokA = (float*)alloc(Mtok * C4 * 4);
  float*    tokB = (float*)alloc(Mtok * C4 * 4);
  uint16_t* cond_bf = (uint16_t*)alloc((size_t)Bn * 512 * 2);
  uint16_t* kc_bf = (uint16_t*)alloc((size_t)Bn * C4 * 2);
  float*    vc_f  = (float*)alloc((size_t)Bn * C4 * 4);
  uint16_t* vct_bf = (uint16_t*)alloc((size_t)C4 * 2);
  float*    finalF = (float*)alloc((size_t)Bn * H0 * W0 * 3 * 4);
  const size_t im2colElems = (size_t)24 * 1024 * 1024;
  uint16_t* colBuf = (uint16_t*)alloc(im2colElems * 2);

  // ---- launch helpers ----
  auto gemm = [&](const uint16_t* A, const uint16_t* Btm, float* Cc, uint16_t* Cb,
                  const float* sc, const float* bi, const float* add,
                  int M, int N, int K, int lda, int ldb, int ldc, float alpha, int flags) {
    dim3 g((unsigned)((N + 127) / 128), (unsigned)((M + 127) / 128));
    k_gemm_bf16<<<g, dim3(256), 0, stream>>>(A, Btm, Cc, Cb, sc, bi, add,
                                             M, N, K, lda, ldb, ldc, alpha, flags);
  };
  auto prepw = [&](const float* src, uint16_t* dst, int O, int I, int KH, int KW, int flip) {
    size_t n = (size_t)O * I * KH * KW;
    k_prep_w<<<(unsigned)((n + 255) / 256), 256, 0, stream>>>(src, dst, O, I, KH, KW, flip);
  };
  auto conv2d = [&](const uint16_t* xbf, int Hc, int Wc, int Cin, int useZ,
                    const uint16_t* Wbf, int Cout, int KH, int KW, int stride, int pad,
                    int reflect, int dil, const float* sc, const float* bi, float alpha,
                    int flags, const float* addsrc, float* outF, uint16_t* outB,
                    int OHc, int OWc) {
    int Kc = (Cin + (useZ ? 16 : 0)) * KH * KW;
    int chunk = (int)(im2colElems / ((size_t)Kc * OWc));
    if (chunk < 1) chunk = 1;
    if (chunk > OHc) chunk = OHc;
    for (int b = 0; b < Bn; b++) {
      for (int oh0 = 0; oh0 < OHc; oh0 += chunk) {
        int ohc = hmin(chunk, OHc - oh0);
        int Mc = ohc * OWc;
        size_t nel = (size_t)Mc * Kc;
        k_im2col<<<(unsigned)((nel + 255) / 256), 256, 0, stream>>>(
            xbf, useZ ? zvec : (const float*)nullptr, colBuf,
            b, Hc, Wc, Cin, useZ ? 16 : 0, KH, KW, stride, pad, dil, reflect,
            OWc, oh0, Mc, Kc);
        size_t obase = ((size_t)b * OHc + oh0) * (size_t)OWc * Cout;
        gemm(colBuf, Wbf, outF ? outF + obase : nullptr, outB ? outB + obase : nullptr,
             sc, bi, addsrc ? addsrc + obase : nullptr,
             Mc, Cout, Kc, Kc, Kc, Cout, alpha, flags);
      }
    }
  };

  // ---- weight prep (bf16, N x K rows) ----
  prepw(c1_w, w_c1, 64, 19, 7, 7, 0);
  prepw(c2_w, w_c2, 128, 80, 3, 3, 0);
  prepw(c3_w, w_c3, 256, 144, 3, 3, 0);
  for (int r = 0; r < 12; r++)
    prepw(res_w + (size_t)r * 256 * 256 * 9, w_res[r], 256, 256, 3, 3, 0);
  for (int i = 0; i < 2; i++) {
    prepw(st_pin_w + (size_t)i * 65536, w_pin[i], 256, 256, 1, 1, 0);
    prepw(st_pout_w + (size_t)i * 65536, w_pout[i], 256, 256, 1, 1, 0);
    prepw(st_q1 + (size_t)i * 65536, w_q1[i], 256, 256, 1, 1, 0);
    prepw(st_k1 + (size_t)i * 65536, w_k1[i], 256, 256, 1, 1, 0);
    prepw(st_v1 + (size_t)i * 65536, w_v1[i], 256, 256, 1, 1, 0);
    prepw(st_o1w + (size_t)i * 65536, w_o1[i], 256, 256, 1, 1, 0);
    prepw(st_q2 + (size_t)i * 65536, w_q2[i], 256, 256, 1, 1, 0);
    prepw(st_k2 + (size_t)i * 131072, w_k2[i], 256, 512, 1, 1, 0);
    prepw(st_v2 + (size_t)i * 131072, w_v2[i], 256, 512, 1, 1, 0);
    prepw(st_o2w + (size_t)i * 65536, w_o2[i], 256, 256, 1, 1, 0);
    prepw(st_ffp_w + (size_t)i * 2048 * 256, w_ffp[i], 2048, 256, 1, 1, 0);
    prepw(st_ffo_w + (size_t)i * 256 * 1024, w_ffo[i], 256, 1024, 1, 1, 0);
  }
  prepw(up1_w, w_up1, 128, 256, 3, 3, 1);   // (256,128,3,3) flipped+swapped
  prepw(up2_w, w_up2, 64, 128, 3, 3, 1);    // (128,64,3,3)
  prepw(cf_w, w_cf, 3, 64, 7, 7, 0);

  // ---- spectral-norm z, cond pack, input layout ----
  k_spectral_z<<<1, 512, 0, stream>>>(in_cond, sn_w, sn_u, zvec);
  k_pack_bf16<<<(unsigned)((Bn * 512 + 255) / 256), 256, 0, stream>>>(in_cond, cond_bf, (size_t)Bn * 512);
  {
    size_t n = (size_t)Bn * 3 * H0 * W0;
    k_nchw_to_nhwc_bf16<<<(unsigned)((n + 255) / 256), 256, 0, stream>>>(in_input, x_in_bf, Bn, 3, H0, W0);
  }

  // ---- downsampling conv stack (BN+ReLU fused into GEMM epilogue) ----
  conv2d(x_in_bf, 224, 224, 3, 1, w_c1, 64, 7, 7, 1, 3, 1, 1,
         bn1_g, bn1_b, bnInv, 1 | 2 | 4, nullptr, nullptr, xbA, 224, 224);
  conv2d(xbA, 224, 224, 64, 1, w_c2, 128, 3, 3, 2, 1, 0, 1,
         bn2_g, bn2_b, bnInv, 1 | 2 | 4, nullptr, nullptr, xbB, 112, 112);
  conv2d(xbB, 112, 112, 128, 1, w_c3, 256, 3, 3, 2, 1, 0, 1,
         bn3_g, bn3_b, bnInv, 1 | 2 | 4, nullptr, xfA, xbA, 56, 56);

  float* curF = xfA; uint16_t* curB = xbA;
  float* altF = xfB; uint16_t* altB = xbB;
  auto swapBuf = [&]() {
    float* tf = curF; curF = altF; altF = tf;
    uint16_t* tb = curB; curB = altB; altB = tb;
  };

  auto resblock = [&](int r) {
    const float* g0 = res_g + (size_t)(r * 2 + 0) * 256;
    const float* b0 = res_b + (size_t)(r * 2 + 0) * 256;
    const float* g1 = res_g + (size_t)(r * 2 + 1) * 256;
    const float* b1 = res_b + (size_t)(r * 2 + 1) * 256;
    conv2d(curB, 56, 56, 256, 0, w_res[2 * r], 256, 3, 3, 1, 1, 1, 1,
           g0, b0, bnInv, 1 | 2 | 4, nullptr, nullptr, h_bf, 56, 56);
    conv2d(h_bf, 56, 56, 256, 0, w_res[2 * r + 1], 256, 3, 3, 1, 1, 1, 1,
           g1, b1, bnInv, 2 | 4 | 8, curF, altF, altB, 56, 56);
    swapBuf();
  };

  auto transformer = [&](int i) {
    float* xF = curF;
    // GroupNorm(32 groups) -> bf16 tokens
    k_groupnorm_bf16<<<(unsigned)(Bn * 32), 256, 0, stream>>>(
        xF, st_gn_g + (size_t)i * 256, st_gn_b + (size_t)i * 256, h_bf, HW, 256, 8);
    // proj_in (1x1 conv == GEMM) + bias
    gemm(h_bf, w_pin[i], tokA, nullptr, nullptr, st_pin_b + (size_t)i * 256, nullptr,
         (int)Mtok, 256, 256, 256, 256, 256, 1.f, 4);
    float* t = tokA; float* t2 = tokB;

    // --- self attention ---
    k_layernorm_bf16<<<(unsigned)Mtok, 256, 0, stream>>>(
        t, st_ln1g + (size_t)i * 256, st_ln1b + (size_t)i * 256, h_bf, 256);
    gemm(h_bf, w_q1[i], nullptr, q_bf, nullptr, nullptr, nullptr, (int)Mtok, 256, 256, 256, 256, 256, 1.f, 0);
    gemm(h_bf, w_k1[i], nullptr, k_bf, nullptr, nullptr, nullptr, (int)Mtok, 256, 256, 256, 256, 256, 1.f, 0);
    gemm(h_bf, w_v1[i], v_f, nullptr, nullptr, nullptr, nullptr, (int)Mtok, 256, 256, 256, 256, 256, 1.f, 0);
    for (int b = 0; b < Bn; b++) {
      size_t off = (size_t)b * HW * 256;
      k_transpose_bf16<<<(unsigned)(((size_t)HW * 256 + 255) / 256), 256, 0, stream>>>(
          v_f + off, vt_bf, HW, 256);
      gemm(q_bf + off, k_bf + off, scoresF, nullptr, nullptr, nullptr, nullptr,
           HW, HW, 256, 256, 256, HW, 0.0625f, 0);                 // 1/sqrt(256)
      k_softmax_bf16<<<(unsigned)HW, 256, 0, stream>>>(scoresF, probsB, HW);
      gemm(probsB, vt_bf, nullptr, o_bf + off, nullptr, nullptr, nullptr,
           HW, 256, HW, HW, HW, 256, 1.f, 0);                      // P @ V
    }
    gemm(o_bf, w_o1[i], t2, nullptr, nullptr, st_o1b + (size_t)i * 256, t,
         (int)Mtok, 256, 256, 256, 256, 256, 1.f, 4 | 8);
    { float* tmp = t; t = t2; t2 = tmp; }

    // --- cross attention (ctx len 1) ---
    k_layernorm_bf16<<<(unsigned)Mtok, 256, 0, stream>>>(
        t, st_ln2g + (size_t)i * 256, st_ln2b + (size_t)i * 256, h_bf, 256);
    gemm(h_bf, w_q2[i], nullptr, q_bf, nullptr, nullptr, nullptr, (int)Mtok, 256, 256, 256, 256, 256, 1.f, 0);
    gemm(cond_bf, w_k2[i], nullptr, kc_bf, nullptr, nullptr, nullptr, Bn, 256, 512, 512, 512, 256, 1.f, 0);
    gemm(cond_bf, w_v2[i], vc_f, nullptr, nullptr, nullptr, nullptr, Bn, 256, 512, 512, 512, 256, 1.f, 0);
    for (int b = 0; b < Bn; b++) {
      size_t off = (size_t)b * HW * 256;
      k_transpose_bf16<<<1, 256, 0, stream>>>(vc_f + (size_t)b * 256, vct_bf, 1, 256);
      gemm(q_bf + off, kc_bf + (size_t)b * 256, scoresF, nullptr, nullptr, nullptr, nullptr,
           HW, 1, 256, 256, 256, 1, 0.0625f, 0);
      k_softmax_bf16<<<(unsigned)HW, 256, 0, stream>>>(scoresF, probsB, 1);
      gemm(probsB, vct_bf, nullptr, o_bf + off, nullptr, nullptr, nullptr,
           HW, 256, 1, 1, 1, 256, 1.f, 0);
    }
    gemm(o_bf, w_o2[i], t2, nullptr, nullptr, st_o2b + (size_t)i * 256, t,
         (int)Mtok, 256, 256, 256, 256, 256, 1.f, 4 | 8);
    { float* tmp = t; t = t2; t2 = tmp; }

    // --- GEGLU feed-forward ---
    k_layernorm_bf16<<<(unsigned)Mtok, 256, 0, stream>>>(
        t, st_ln3g + (size_t)i * 256, st_ln3b + (size_t)i * 256, h_bf, 256);
    gemm(h_bf, w_ffp[i], pF, nullptr, nullptr, st_ffp_b + (size_t)i * 2048, nullptr,
         (int)Mtok, 2048, 256, 256, 256, 2048, 1.f, 4);
    k_geglu_bf16<<<(unsigned)((Mtok * 1024 + 255) / 256), 256, 0, stream>>>(pF, g_bf, Mtok, 1024);
    gemm(g_bf, w_ffo[i], t2, nullptr, nullptr, st_ffo_b + (size_t)i * 256, t,
         (int)Mtok, 256, 1024, 1024, 1024, 256, 1.f, 4 | 8);
    { float* tmp = t; t = t2; t2 = tmp; }

    // proj_out + residual with pre-transformer activations
    k_pack_bf16<<<(unsigned)((Mtok * 256 + 255) / 256), 256, 0, stream>>>(t, t_bf, Mtok * 256);
    gemm(t_bf, w_pout[i], altF, altB, nullptr, st_pout_b + (size_t)i * 256, xF,
         (int)Mtok, 256, 256, 256, 256, 256, 1.f, 4 | 8);
    swapBuf();
  };

  resblock(0); resblock(1);
  transformer(0);
  resblock(2); resblock(3);
  transformer(1);
  resblock(4); resblock(5);

  // ---- upsampling (transposed convs as input-dilated convs) ----
  conv2d(curB, 56, 56, 256, 0, w_up1, 128, 3, 3, 1, 1, 0, 2,
         up1_g, up1_b, bnInv, 1 | 2 | 4, nullptr, nullptr, altB, 112, 112);
  swapBuf();
  conv2d(curB, 112, 112, 128, 0, w_up2, 64, 3, 3, 1, 1, 0, 2,
         up2_g, up2_b, bnInv, 1 | 2 | 4, nullptr, nullptr, altB, 224, 224);
  swapBuf();
  conv2d(curB, 224, 224, 64, 0, w_cf, 3, 7, 7, 1, 3, 1, 1,
         nullptr, cf_b, 1.f, 4, nullptr, finalF, nullptr, 224, 224);

  // ---- tanh -> 3x3 gaussian blur -> *eps -> NCHW output ----
  {
    size_t n = (size_t)Bn * 3 * H0 * W0;
    k_tanh_blur<<<(unsigned)((n + 255) / 256), 256, 0, stream>>>(finalF, (float*)d_out, Bn, H0, W0, epsp);
  }
}
